// EncoderLayer_45715631898652
// MI455X (gfx1250) — matrix operand order repaired and checked by reading, not compiled, not run on hardware
//
#include <hip/hip_runtime.h>
#include <hip/hip_bf16.h>

#define BB  2
#define SS  4096
#define EE  512
#define HH  8
#define DDm 64
#define FFF 2048
#define MM  (BB * SS)

typedef __attribute__((ext_vector_type(16))) _Float16     v16h;
typedef __attribute__((ext_vector_type(8)))  float        v8f;
typedef __attribute__((ext_vector_type(4)))  unsigned int u32x4;

typedef __attribute__((ext_vector_type(4))) unsigned int tdm_g0_t;
typedef __attribute__((ext_vector_type(8))) int          tdm_g1_t;
typedef __attribute__((ext_vector_type(4))) int          tdm_g2_t;

union FragH {
    v16h  h;
    u32x4 q[2];
};

__device__ __forceinline__ v8f wmma_f16(v16h a, v16h b, v8f c) {
    // 8 args: (neg_a, A, neg_b, B, c_mod, C, reuse_a, reuse_b)
    return __builtin_amdgcn_wmma_f32_16x16x32_f16(false, a, false, b, (short)0, c,
                                                  false, false);
}

// ---------------------------------------------------------------------------
// TDM: 2-D tile load (tile_dim0 x tile_dim1 elements of 2 bytes) into LDS.
// D# per CDNA5 ISA ch.8: group0 = {count, lds_addr, global_addr, type=2},
// group1 = {data_size, tensor dims, tile dims, dim0 stride}. Groups 2/3 zero
// (2-D tensor). Issued per-wave; tracked by TENSORcnt.
// ---------------------------------------------------------------------------
__device__ __forceinline__ void tdm_load_2d(const _Float16* gptr, unsigned lds_off,
                                            int tensor_d0, int tensor_d1,
                                            int tile_d0, int tile_d1) {
    unsigned long long ga = (unsigned long long)(const void*)gptr;
    tdm_g0_t g0;
    g0[0] = 1u;                                               // count=1, user mode
    g0[1] = lds_off;                                          // lds_addr (bytes)
    g0[2] = (unsigned)(ga & 0xFFFFFFFFu);                     // global_addr[31:0]
    g0[3] = (unsigned)((ga >> 32) & 0x1FFFFFFu) | (2u << 30); // addr[56:32] | type=2

    tdm_g1_t g1;
    g1[0] = 1 << 16;                                          // wg_mask=0, data_size=2B
    g1[1] = (int)(((unsigned)tensor_d0 & 0xFFFFu) << 16);     // tensor_dim0[15:0]
    g1[2] = (int)((((unsigned)tensor_d0 >> 16) & 0xFFFFu) |
                  (((unsigned)tensor_d1 & 0xFFFFu) << 16));   // td0 hi | td1 lo
    g1[3] = (int)((((unsigned)tensor_d1 >> 16) & 0xFFFFu) |
                  (((unsigned)tile_d0 & 0xFFFFu) << 16));     // td1 hi | tile_dim0
    g1[4] = (int)((unsigned)tile_d1 & 0xFFFFu);               // tile_dim1, tile_dim2=0
    g1[5] = tensor_d0;                                        // dim0 stride [31:0]
    g1[6] = 0;                                                // stride0 hi, stride1 lo
    g1[7] = 0;

    tdm_g2_t z4 = {};
#if defined(__clang_major__) && (__clang_major__ >= 23)
    tdm_g1_t z8 = {};
    __builtin_amdgcn_tensor_load_to_lds(g0, g1, z4, z4, z8, 0);
#else
    __builtin_amdgcn_tensor_load_to_lds(g0, g1, z4, z4, 0);
#endif
}

// ---------------------------------------------------------------------------
// Weight transpose + f32->f16 convert:  W[K][N] (f32) -> WT[N][K] (f16)
// ---------------------------------------------------------------------------
__global__ void convert_wt(const float* __restrict__ W, _Float16* __restrict__ WT,
                           int K, int N) {
    size_t idx = (size_t)blockIdx.x * blockDim.x + threadIdx.x;
    if (idx >= (size_t)K * N) return;
    int k = (int)(idx / N);
    int n = (int)(idx % N);
    WT[(size_t)n * K + k] = (_Float16)W[idx];
}

// ---------------------------------------------------------------------------
// LayerNorm (torch-style: unbiased std, scalar alpha/bias) f32 -> f16
// One block (256 thr) per row of E=512.
// ---------------------------------------------------------------------------
__global__ void ln_kernel(const float* __restrict__ x, const float* __restrict__ alpha_p,
                          const float* __restrict__ bias_p, _Float16* __restrict__ out) {
    const int E = EE;
    int row = blockIdx.x;
    int tid = threadIdx.x;
    const float* xr = x + (size_t)row * E;
    float v0 = xr[tid];
    float v1 = xr[tid + 256];
    float s  = v0 + v1;
    float s2 = v0 * v0 + v1 * v1;
#pragma unroll
    for (int m = 1; m < 32; m <<= 1) {
        s  += __shfl_xor(s,  m, 32);
        s2 += __shfl_xor(s2, m, 32);
    }
    __shared__ float red[2][8];
    int wave = tid >> 5, lane = tid & 31;
    if (lane == 0) { red[0][wave] = s; red[1][wave] = s2; }
    __syncthreads();
    if (wave == 0) {
        float a = (lane < 8) ? red[0][lane] : 0.f;
        float b = (lane < 8) ? red[1][lane] : 0.f;
#pragma unroll
        for (int m = 1; m < 8; m <<= 1) {
            a += __shfl_xor(a, m, 32);
            b += __shfl_xor(b, m, 32);
        }
        if (lane == 0) { red[0][0] = a; red[1][0] = b; }
    }
    __syncthreads();
    float sum   = red[0][0];
    float sumsq = red[1][0];
    float mean  = sum / (float)E;
    float var   = (sumsq - (float)E * mean * mean) / (float)(E - 1);
    float stdv  = sqrtf(fmaxf(var, 0.f));
    float scale = alpha_p[0] / (stdv + 1e-6f);
    float bb    = bias_p[0];
    out[(size_t)row * E + tid]       = (_Float16)((v0 - mean) * scale + bb);
    out[(size_t)row * E + tid + 256] = (_Float16)((v1 - mean) * scale + bb);
}

// ---------------------------------------------------------------------------
// WMMA GEMM: out = A[M,K](f16) @ W[K,N] + bias, W given as WT[N][K] (f16).
// 256 thr = 8 waves; wave computes 16(M) x 64(N); grid = (N/64, M/128).
// The shared 64x32 WT tile is TDM-staged into LDS (double buffered): one DMA
// per block per k-step instead of 8 redundant per-wave global fetches.
// EPI: 0 = f16 row-major, 1 = f16 V-transpose [B,H,D,S],
//      2 = f32 + residual,  3 = relu -> f16
// ---------------------------------------------------------------------------
template <int EPI>
__global__ void gemm16(const _Float16* __restrict__ A, const _Float16* __restrict__ WT,
                       const float* __restrict__ bias, const float* __restrict__ resid,
                       void* __restrict__ outp, int M, int N, int K) {
    __shared__ __align__(16) _Float16 bstage[2][64 * 32];  // [buf][col][32 k halves]

    int lane = threadIdx.x & 31;
    int wave = threadIdx.x >> 5;
    int hf   = lane >> 4;
    int ln   = lane & 15;
    int row0 = blockIdx.y * 128 + wave * 16;
    int col0 = blockIdx.x * 64;

    const _Float16* arow  = A + (size_t)(row0 + ln) * K;
    const _Float16* wbase = WT + (size_t)col0 * K;
    v8f acc[4] = {};

    // prologue: stage first tile
    if (wave == 0) {
        tdm_load_2d(wbase, (unsigned)(size_t)&bstage[0][0], K, N, 32, 64);
        __builtin_amdgcn_s_wait_tensorcnt(0);
    }
    __syncthreads();

    for (int k0 = 0; k0 < K; k0 += 32) {
        int  cb   = (k0 >> 5) & 1;
        bool more = (k0 + 32) < K;
        if (wave == 0 && more)  // overlap next DMA with this tile's WMMAs
            tdm_load_2d(wbase + (k0 + 32), (unsigned)(size_t)&bstage[cb ^ 1][0],
                        K, N, 32, 64);

        FragH a;
        a.q[0] = *(const u32x4*)(arow + k0 + 8 * hf);
        a.q[1] = *(const u32x4*)(arow + k0 + 8 * hf + 16);
#pragma unroll
        for (int c = 0; c < 4; ++c) {
            const _Float16* bp = &bstage[cb][(c * 16 + ln) * 32 + 8 * hf];
            FragH b;
            b.q[0] = *(const u32x4*)(bp);
            b.q[1] = *(const u32x4*)(bp + 16);
            acc[c] = wmma_f16(a.h, b.h, acc[c]);
        }

        if (wave == 0 && more) __builtin_amdgcn_s_wait_tensorcnt(0);
        __syncthreads();
    }

#pragma unroll
    for (int c = 0; c < 4; ++c) {
        int   n  = col0 + c * 16 + ln;
        float bv = bias[n];
#pragma unroll
        for (int i = 0; i < 8; ++i) {
            int   r = row0 + 8 * hf + i;
            float v = acc[c][i] + bv;
            if constexpr (EPI == 0) {
                ((_Float16*)outp)[(size_t)r * N + n] = (_Float16)v;
            } else if constexpr (EPI == 1) {
                int bidx = r / SS, sidx = r % SS;
                int hd = n >> 6, dd = n & 63;
                ((_Float16*)outp)[(((size_t)bidx * HH + hd) * DDm + dd) * SS + sidx] =
                    (_Float16)v;
            } else if constexpr (EPI == 2) {
                ((float*)outp)[(size_t)r * N + n] = v + resid[(size_t)r * N + n];
            } else {
                ((_Float16*)outp)[(size_t)r * N + n] = (_Float16)(v > 0.f ? v : 0.f);
            }
        }
    }
}

// ---------------------------------------------------------------------------
// Flash attention. Q,K: f16 [B,S,H,D]; VT: f16 [B,H,D,S]; ctx: f16 [B,S,H,D].
// 256 thr = 8 waves; wave handles 16 query rows; loops keys in blocks of 32.
// ---------------------------------------------------------------------------
__global__ void flash_attn(const _Float16* __restrict__ Q, const _Float16* __restrict__ Km,
                           const _Float16* __restrict__ VT, const int* __restrict__ mask,
                           _Float16* __restrict__ ctx) {
    __shared__ __align__(16) _Float16 smem[8][16 * 32];

    int lane = threadIdx.x & 31;
    int wave = threadIdx.x >> 5;
    int hf   = lane >> 4;
    int ln   = lane & 15;

    const int qchunks = SS / 128;
    int bh = blockIdx.x / qchunks;
    int qc = blockIdx.x % qchunks;
    int b  = bh / HH;
    int h  = bh % HH;
    int q0 = qc * 128 + wave * 16;

    // Q A-fragments for the two 32-wide K-chunks of D=64 (loop invariant)
    const _Float16* qrow = Q + ((size_t)b * SS + q0 + ln) * EE + h * DDm;
    FragH aq[2];
#pragma unroll
    for (int kc = 0; kc < 2; ++kc) {
        aq[kc].q[0] = *(const u32x4*)(qrow + kc * 32 + 8 * hf);
        aq[kc].q[1] = *(const u32x4*)(qrow + kc * 32 + 8 * hf + 16);
    }

    float m_run[8], l_run[8];
    v8f   o[4] = {};
#pragma unroll
    for (int i = 0; i < 8; ++i) { m_run[i] = -1e30f; l_run[i] = 0.f; }

    const int* maskb = mask + (size_t)b * SS;

    for (int j0 = 0; j0 < SS; j0 += 32) {
        // --- scores: two 16x16 tiles covering 32 keys ---
        v8f sfr[2];
#pragma unroll
        for (int t = 0; t < 2; ++t) {
            int key = j0 + t * 16 + ln;
            const _Float16* krow = Km + ((size_t)b * SS + key) * EE + h * DDm;
            v8f s = {};
#pragma unroll
            for (int kc = 0; kc < 2; ++kc) {
                FragH bk;
                bk.q[0] = *(const u32x4*)(krow + kc * 32 + 8 * hf);
                bk.q[1] = *(const u32x4*)(krow + kc * 32 + 8 * hf + 16);
                s = wmma_f16(aq[kc].h, bk.h, s);
            }
            bool mz = (maskb[key] == 0);
#pragma unroll
            for (int i = 0; i < 8; ++i) {
                float v = s[i] * 0.125f;  // 1/sqrt(64)
                s[i] = mz ? -1e30f : v;
            }
            sfr[t] = s;
        }

        // --- online softmax statistics (row = register index, col = lane) ---
        float alpha[8];
#pragma unroll
        for (int i = 0; i < 8; ++i) {
            float mx = fmaxf(sfr[0][i], sfr[1][i]);
            mx = fmaxf(mx, __shfl_xor(mx, 1, 32));
            mx = fmaxf(mx, __shfl_xor(mx, 2, 32));
            mx = fmaxf(mx, __shfl_xor(mx, 4, 32));
            mx = fmaxf(mx, __shfl_xor(mx, 8, 32));
            float mn = fmaxf(m_run[i], mx);
            alpha[i] = __expf(m_run[i] - mn);
            float p0 = __expf(sfr[0][i] - mn);
            float p1 = __expf(sfr[1][i] - mn);
            sfr[0][i] = p0;
            sfr[1][i] = p1;
            float ssum = p0 + p1;
            ssum += __shfl_xor(ssum, 1, 32);
            ssum += __shfl_xor(ssum, 2, 32);
            ssum += __shfl_xor(ssum, 4, 32);
            ssum += __shfl_xor(ssum, 8, 32);
            l_run[i] = l_run[i] * alpha[i] + ssum;
            m_run[i] = mn;
        }

        // --- P: C-layout -> A-layout via per-wave LDS bounce ---
#pragma unroll
        for (int t = 0; t < 2; ++t)
#pragma unroll
            for (int i = 0; i < 8; ++i)
                smem[wave][(8 * hf + i) * 32 + t * 16 + ln] = (_Float16)sfr[t][i];
        __asm__ volatile("s_wait_dscnt 0x0" ::: "memory");
        FragH pa;
        pa.q[0] = *(const u32x4*)(&smem[wave][ln * 32 + 8 * hf]);
        pa.q[1] = *(const u32x4*)(&smem[wave][ln * 32 + 8 * hf + 16]);

        // --- O = O*alpha + P @ V  (4 tiles of 16 output columns) ---
#pragma unroll
        for (int nc = 0; nc < 4; ++nc) {
#pragma unroll
            for (int i = 0; i < 8; ++i) o[nc][i] *= alpha[i];
            const _Float16* vrow =
                VT + (((size_t)b * HH + h) * DDm + nc * 16 + ln) * SS + j0 + 8 * hf;
            FragH bv;
            bv.q[0] = *(const u32x4*)(vrow);
            bv.q[1] = *(const u32x4*)(vrow + 16);
            o[nc] = wmma_f16(pa.h, bv.h, o[nc]);
        }
    }

    // --- finalize: divide by row sums, write ctx [B,S,H,D] ---
    float inv[8];
#pragma unroll
    for (int i = 0; i < 8; ++i) inv[i] = 1.f / l_run[i];
#pragma unroll
    for (int nc = 0; nc < 4; ++nc)
#pragma unroll
        for (int i = 0; i < 8; ++i)
            ctx[((size_t)b * SS + q0 + 8 * hf + i) * EE + h * DDm + nc * 16 + ln] =
                (_Float16)(o[nc][i] * inv[i]);
}

// ---------------------------------------------------------------------------
extern "C" void kernel_launch(void* const* d_in, const int* in_sizes, int n_in,
                              void* d_out, int out_size, void* d_ws, size_t ws_size,
                              hipStream_t stream) {
    (void)in_sizes; (void)n_in; (void)out_size; (void)ws_size;

    const float* x    = (const float*)d_in[0];
    const int*   mask = (const int*)d_in[1];
    const float* Wq = (const float*)d_in[2];  const float* bq = (const float*)d_in[3];
    const float* Wk = (const float*)d_in[4];  const float* bk = (const float*)d_in[5];
    const float* Wv = (const float*)d_in[6];  const float* bv = (const float*)d_in[7];
    const float* Wo = (const float*)d_in[8];  const float* bo = (const float*)d_in[9];
    const float* W1 = (const float*)d_in[10]; const float* b1 = (const float*)d_in[11];
    const float* W2 = (const float*)d_in[12]; const float* b2 = (const float*)d_in[13];
    const float* a1 = (const float*)d_in[14]; const float* c1 = (const float*)d_in[15];
    const float* a2 = (const float*)d_in[16]; const float* c2 = (const float*)d_in[17];

    char* p = (char*)d_ws;
    _Float16* WTq = (_Float16*)p; p += (size_t)EE * EE * 2;
    _Float16* WTk = (_Float16*)p; p += (size_t)EE * EE * 2;
    _Float16* WTv = (_Float16*)p; p += (size_t)EE * EE * 2;
    _Float16* WTo = (_Float16*)p; p += (size_t)EE * EE * 2;
    _Float16* WT1 = (_Float16*)p; p += (size_t)EE * FFF * 2;
    _Float16* WT2 = (_Float16*)p; p += (size_t)FFF * EE * 2;
    _Float16* xn  = (_Float16*)p; p += (size_t)MM * EE * 2;
    _Float16* Qh  = (_Float16*)p; p += (size_t)MM * EE * 2;
    _Float16* Kh  = (_Float16*)p; p += (size_t)MM * EE * 2;
    _Float16* VTv = (_Float16*)p; p += (size_t)MM * EE * 2;
    _Float16* ctx = (_Float16*)p; p += (size_t)MM * EE * 2;
    _Float16* xn2 = (_Float16*)p; p += (size_t)MM * EE * 2;
    _Float16* hff = (_Float16*)p; p += (size_t)MM * FFF * 2;
    float*    x1  = (float*)p;    p += (size_t)MM * EE * 4;

    // weights -> transposed f16
    convert_wt<<<(EE * EE + 255) / 256, 256, 0, stream>>>(Wq, WTq, EE, EE);
    convert_wt<<<(EE * EE + 255) / 256, 256, 0, stream>>>(Wk, WTk, EE, EE);
    convert_wt<<<(EE * EE + 255) / 256, 256, 0, stream>>>(Wv, WTv, EE, EE);
    convert_wt<<<(EE * EE + 255) / 256, 256, 0, stream>>>(Wo, WTo, EE, EE);
    convert_wt<<<(EE * FFF + 255) / 256, 256, 0, stream>>>(W1, WT1, EE, FFF);
    convert_wt<<<(FFF * EE + 255) / 256, 256, 0, stream>>>(W2, WT2, FFF, EE);

    // LN1
    ln_kernel<<<MM, 256, 0, stream>>>(x, a1, c1, xn);

    // QKV projections
    dim3 gE(EE / 64, MM / 128);
    gemm16<0><<<gE, 256, 0, stream>>>(xn, WTq, bq, nullptr, (void*)Qh, MM, EE, EE);
    gemm16<0><<<gE, 256, 0, stream>>>(xn, WTk, bk, nullptr, (void*)Kh, MM, EE, EE);
    gemm16<1><<<gE, 256, 0, stream>>>(xn, WTv, bv, nullptr, (void*)VTv, MM, EE, EE);

    // flash attention
    flash_attn<<<BB * HH * (SS / 128), 256, 0, stream>>>(Qh, Kh, VTv, mask, ctx);

    // output projection + residual 1 (f32)
    gemm16<2><<<gE, 256, 0, stream>>>(ctx, WTo, bo, x, (void*)x1, MM, EE, EE);

    // LN2
    ln_kernel<<<MM, 256, 0, stream>>>(x1, a2, c2, xn2);

    // FFN
    dim3 gF(FFF / 64, MM / 128);
    gemm16<3><<<gF, 256, 0, stream>>>(xn2, WT1, b1, nullptr, (void*)hff, MM, FFF, EE);
    gemm16<2><<<gE, 256, 0, stream>>>(hff, WT2, b2, x1, d_out, MM, EE, FFF);
}
